// MultiHeadAttention_16011638079994
// MI455X (gfx1250) — matrix operand order repaired and checked by reading, not compiled, not run on hardware
//
#include <hip/hip_runtime.h>

typedef _Float16 half_t;
typedef _Float16 v8h  __attribute__((ext_vector_type(8)));
typedef _Float16 v16h __attribute__((ext_vector_type(16)));
typedef float    v8f  __attribute__((ext_vector_type(8)));

#define NUM_HEADS 16
#define HEAD_DIM  64
#define HIDDEN    1024
#define BATCH     2
#define SEQ       2048
#define MTOT      (BATCH * SEQ)   // 4096
#define KSTEP     64              // GEMM K chunk staged in LDS

// ---- workspace layout (bytes) ----
#define XH_OFF   0u                       // 4096*1024 f16 = 8 MB
#define WQH_OFF  (XH_OFF  + 8388608u)     // 1024*1024 f16 = 2 MB
#define WKH_OFF  (WQH_OFF + 2097152u)
#define WVH_OFF  (WKH_OFF + 2097152u)
#define QH_OFF   (WVH_OFF + 2097152u)     // [32][2048][64] f16 = 8 MB
#define KH_OFF   (QH_OFF  + 8388608u)     // same
#define VT_OFF   (KH_OFF  + 8388608u)     // [32][64][2048] f16 = 8 MB (transposed)
#define COS_OFF  (VT_OFF  + 8388608u)     // [2048][64] f32 = 512 KB
#define SIN_OFF  (COS_OFF + 524288u)

// ---------------- CDNA5 async copy: global -> LDS (ASYNCcnt tracked) ----------------
// Inline asm: bypasses the builtin's language-address-space parameter types.
// vdst = VGPR with LDS byte address (low 32 bits of a generic __shared__ pointer),
// vaddr = 64-bit global address, saddr = off.
__device__ __forceinline__ void async_copy_b128(const half_t* g, half_t* l) {
    uint32_t laddr = (uint32_t)(uintptr_t)l;
    uint64_t gaddr = (uint64_t)(uintptr_t)g;
    asm volatile("global_load_async_to_lds_b128 %0, %1, off"
                 :: "v"(laddr), "v"(gaddr) : "memory");
}

__device__ __forceinline__ void wait_async0() {
    asm volatile("s_wait_asynccnt 0" ::: "memory");
}

// ---------------- f32 -> f16 convert ----------------
__global__ void cvt_f32_f16(const float* __restrict__ src, half_t* __restrict__ dst, int n) {
    int i = blockIdx.x * blockDim.x + threadIdx.x;
    if (i < n) dst[i] = (half_t)src[i];
}

// ---------------- RoPE cos/sin tables ----------------
__global__ void rope_tables(float* __restrict__ cosT, float* __restrict__ sinT) {
    int i = blockIdx.x * blockDim.x + threadIdx.x;   // SEQ*HEAD_DIM
    int s = i >> 6, d = i & 63;
    int j = d & 31;                                  // freq index
    float invf = __expf(-(float)j * (9.210340371976184f / 32.0f));  // 10000^(-2j/64)
    float ang  = (float)s * invf;
    cosT[i] = __cosf(ang);
    sinT[i] = __sinf(ang);
}

// ---------------- fused QKV projection (WMMA) + RoPE epilogue ----------------
// out[m,n] = sum_k X[m,k] * W[n,k]. Block: 8 waves = 128 M x 64 N (one head).
// B tile (64 rows x KSTEP halves = 8 KB) staged via async DMA, double buffered.
__global__ __launch_bounds__(256)
void qkv_gemm(const half_t* __restrict__ X, const half_t* __restrict__ W,
              const float* __restrict__ cosT, const float* __restrict__ sinT,
              half_t* __restrict__ Orm,   // rope'd, row-major [bh][s][64]  (Q/K)
              half_t* __restrict__ Otr,   // transposed      [bh][64][s]   (V)
              int do_rope) {
    __shared__ half_t Bl[2][64 * KSTEP];             // 2 x 8 KB

    const int tid  = threadIdx.x;
    const int wave = tid >> 5;
    const int lane = tid & 31;
    const int lh = lane & 15;
    const int hi = lane >> 4;

    const int m0 = blockIdx.x * 128 + wave * 16;
    const int n0 = blockIdx.y * 64;

    // staging: thread t -> W row n0 + (t>>2), halves (t&3)*16 .. +15 (32 B = 2 x b128)
    const int srow = tid >> 2, sq = tid & 3;
    const half_t* wbase = W + (size_t)(n0 + srow) * HIDDEN + sq * 16;

    // prefetch first chunk
    async_copy_b128(wbase + 0, &Bl[0][srow * KSTEP + sq * 16]);
    async_copy_b128(wbase + 8, &Bl[0][srow * KSTEP + sq * 16 + 8]);

    const half_t* aptr = X + (size_t)(m0 + lh) * HIDDEN + hi * 16;

    v8f acc[4] = {};
    for (int it = 0; it < HIDDEN / KSTEP; ++it) {
        const int kk  = it * KSTEP;
        const int cur = it & 1;

        wait_async0();          // my DMA writes for Bl[cur] done
        __syncthreads();        // everyone's done; everyone finished reading Bl[cur^1]

        if (kk + KSTEP < HIDDEN) {
            const half_t* ws = wbase + kk + KSTEP;
            async_copy_b128(ws + 0, &Bl[cur ^ 1][srow * KSTEP + sq * 16]);
            async_copy_b128(ws + 8, &Bl[cur ^ 1][srow * KSTEP + sq * 16 + 8]);
        }
        __builtin_prefetch(aptr + kk + 2 * KSTEP, 0, 0);   // A stream hint

        const half_t* bl = &Bl[cur][0];
#pragma unroll
        for (int ks = 0; ks < KSTEP; ks += 32) {
            // A fragment: lane -> row m0+lh ; K = kk+ks + hi*16 + 0..15
            v8h alo = *(const v8h*)(aptr + kk + ks);
            v8h ahi = *(const v8h*)(aptr + kk + ks + 8);
            v16h a;
#pragma unroll
            for (int i = 0; i < 8; ++i) { a[i] = alo[i]; a[8 + i] = ahi[i]; }
#pragma unroll
            for (int j = 0; j < 4; ++j) {
                // B fragment from LDS: col n0+16j+lh, K = ks + hi*16 + 0..15
                v16h b = *(const v16h*)(bl + (16 * j + lh) * KSTEP + ks + hi * 16);
                acc[j] = __builtin_amdgcn_wmma_f32_16x16x32_f16(
                    false, a, false, b, (short)0, acc[j], false, false);
            }
        }
    }

    // epilogue: C layout -> lane holds col n, rows m0 + r + 8*hi
    const int h  = n0 >> 6;
    const int b  = m0 / SEQ;
    const int s0 = m0 % SEQ;
    const int bh = b * NUM_HEADS + h;

    if (do_rope) {
#pragma unroll
        for (int j = 0; j < 4; ++j) {
            const int d = 16 * j + lh;
            const int pj = j ^ 2;         // RoPE partner accumulator (d +/- 32)
#pragma unroll
            for (int r = 0; r < 8; ++r) {
                const int s = s0 + r + 8 * hi;
                float cs = cosT[s * HEAD_DIM + d];
                float sn = sinT[s * HEAD_DIM + d];
                float x  = acc[j][r];
                float xr = (j < 2) ? -acc[pj][r] : acc[pj][r];
                Orm[((size_t)bh * SEQ + s) * HEAD_DIM + d] = (half_t)(x * cs + xr * sn);
            }
        }
    } else {
        // V: store transposed [bh][d][s]; lane's 8 rows contiguous in s
#pragma unroll
        for (int j = 0; j < 4; ++j) {
            const int d = 16 * j + lh;
            v8h pk;
#pragma unroll
            for (int r = 0; r < 8; ++r) pk[r] = (half_t)acc[j][r];
            *(v8h*)(Otr + ((size_t)bh * HEAD_DIM + d) * SEQ + s0 + 8 * hi) = pk;
        }
    }
}

// ---------------- flash attention (WMMA + async K/V staging + online softmax) ----------------
// Block: 4 waves, same head; each wave owns a 16-row Q tile. Per 32-key chunk the
// block async-stages K (32x64) and V^T (64x32) tiles into double-buffered LDS.
__global__ __launch_bounds__(128)
void flash_attn(const half_t* __restrict__ Qh, const half_t* __restrict__ Kh,
                const half_t* __restrict__ Vt, float* __restrict__ out) {
    __shared__ half_t Kl[2][32 * 64];     // 2 x 4 KB
    __shared__ half_t Vl[2][64 * 32];     // 2 x 4 KB
    __shared__ half_t Pbuf[4][16 * 32];   // per-wave P bounce, 4 KB

    const int tid  = threadIdx.x;
    const int wave = tid >> 5;
    const int lane = tid & 31;
    const int lh = lane & 15;
    const int hi = lane >> 4;

    const int bh = blockIdx.y;
    const int q0 = (blockIdx.x * 4 + wave) * 16;

    const half_t* Qp = Qh + (size_t)bh * SEQ * HEAD_DIM;
    const half_t* Kp = Kh + (size_t)bh * SEQ * HEAD_DIM;
    const half_t* Vp = Vt + (size_t)bh * HEAD_DIM * SEQ;
    half_t* P = &Pbuf[wave][0];

    // staging assignments
    const int krow = tid >> 2, kq = tid & 3;   // K: row 0..31, quarter 0..3
    const int vd = tid >> 1, vh = tid & 1;     // V: d 0..63, half 0..1

    // prefetch chunk 0
    {
        const half_t* ks = Kp + (size_t)krow * HEAD_DIM + kq * 16;
        async_copy_b128(ks + 0, &Kl[0][krow * 64 + kq * 16]);
        async_copy_b128(ks + 8, &Kl[0][krow * 64 + kq * 16 + 8]);
        const half_t* vs = Vp + (size_t)vd * SEQ + vh * 16;
        async_copy_b128(vs + 0, &Vl[0][vd * 32 + vh * 16]);
        async_copy_b128(vs + 8, &Vl[0][vd * 32 + vh * 16 + 8]);
    }

    // Q A-fragments for the two K-dim chunks (d 0..31, 32..63)
    v16h qa[2];
    {
        const half_t* qptr = Qp + (size_t)(q0 + lh) * HEAD_DIM + hi * 16;
#pragma unroll
        for (int kc = 0; kc < 2; ++kc) {
            v8h lo  = *(const v8h*)(qptr + 32 * kc);
            v8h hi8 = *(const v8h*)(qptr + 32 * kc + 8);
#pragma unroll
            for (int i = 0; i < 8; ++i) { qa[kc][i] = lo[i]; qa[kc][8 + i] = hi8[i]; }
        }
    }

    float mrow[8], lrow[8];
#pragma unroll
    for (int r = 0; r < 8; ++r) { mrow[r] = -1e30f; lrow[r] = 0.0f; }
    v8f octx[4] = {};

    const float scale = 0.125f;           // 1/sqrt(64)

    for (int it = 0; it < SEQ / 32; ++it) {
        const int key0 = it * 32;
        const int cur  = it & 1;

        wait_async0();
        __syncthreads();

        if (key0 + 32 < SEQ) {
            const half_t* ks = Kp + (size_t)(key0 + 32 + krow) * HEAD_DIM + kq * 16;
            async_copy_b128(ks + 0, &Kl[cur ^ 1][krow * 64 + kq * 16]);
            async_copy_b128(ks + 8, &Kl[cur ^ 1][krow * 64 + kq * 16 + 8]);
            const half_t* vs = Vp + (size_t)vd * SEQ + key0 + 32 + vh * 16;
            async_copy_b128(vs + 0, &Vl[cur ^ 1][vd * 32 + vh * 16]);
            async_copy_b128(vs + 8, &Vl[cur ^ 1][vd * 32 + vh * 16 + 8]);
        }

        // ---- scores: two 16-key tiles, K-dim = 64 -> 2 WMMAs each ----
        v8f sc[2] = {};
#pragma unroll
        for (int t = 0; t < 2; ++t) {
#pragma unroll
            for (int kc = 0; kc < 2; ++kc) {
                v16h kb = *(const v16h*)(&Kl[cur][(16 * t + lh) * 64 + 32 * kc + hi * 16]);
                sc[t] = __builtin_amdgcn_wmma_f32_16x16x32_f16(
                    false, qa[kc], false, kb, (short)0, sc[t], false, false);
            }
        }

        // ---- online softmax; each row spans the 16 lanes of a half-wave ----
#pragma unroll
        for (int r = 0; r < 8; ++r) {
            float v0 = sc[0][r] * scale;
            float v1 = sc[1][r] * scale;
            float mx = fmaxf(v0, v1);
            mx = fmaxf(mx, __shfl_xor(mx, 8, 16));
            mx = fmaxf(mx, __shfl_xor(mx, 4, 16));
            mx = fmaxf(mx, __shfl_xor(mx, 2, 16));
            mx = fmaxf(mx, __shfl_xor(mx, 1, 16));
            float mnew  = fmaxf(mrow[r], mx);
            float alpha = __expf(mrow[r] - mnew);
            float p0 = __expf(v0 - mnew);
            float p1 = __expf(v1 - mnew);
            float ps = p0 + p1;
            ps += __shfl_xor(ps, 8, 16);
            ps += __shfl_xor(ps, 4, 16);
            ps += __shfl_xor(ps, 2, 16);
            ps += __shfl_xor(ps, 1, 16);
            lrow[r] = lrow[r] * alpha + ps;
            mrow[r] = mnew;
#pragma unroll
            for (int j = 0; j < 4; ++j) octx[j][r] *= alpha;
            const int row = r + 8 * hi;
            P[row * 32 + lh]      = (half_t)p0;
            P[row * 32 + 16 + lh] = (half_t)p1;
        }

        // wave-private LDS turnaround (DS in-order per wave; stop compiler reordering)
        asm volatile("s_wait_dscnt 0" ::: "memory");

        // ---- reload P as A fragment [16 x 32] ----
        v16h pa;
        {
            const half_t* pptr = P + lh * 32 + hi * 16;
            v8h lo  = *(const v8h*)(pptr);
            v8h hi8 = *(const v8h*)(pptr + 8);
#pragma unroll
            for (int i = 0; i < 8; ++i) { pa[i] = lo[i]; pa[8 + i] = hi8[i]; }
        }

        // ---- ctx += P x V from staged transposed-V tile ----
#pragma unroll
        for (int j = 0; j < 4; ++j) {
            v16h vb = *(const v16h*)(&Vl[cur][(16 * j + lh) * 32 + hi * 16]);
            octx[j] = __builtin_amdgcn_wmma_f32_16x16x32_f16(
                false, pa, false, vb, (short)0, octx[j], false, false);
        }

        asm volatile("s_wait_dscnt 0" ::: "memory");
    }

    // ---- epilogue: normalize and scatter to [B, S, H*D] f32 ----
    const int b = bh >> 4;
    const int h = bh & 15;
#pragma unroll
    for (int j = 0; j < 4; ++j) {
        const int d = h * HEAD_DIM + 16 * j + lh;
#pragma unroll
        for (int r = 0; r < 8; ++r) {
            const int s = q0 + r + 8 * hi;
            out[((size_t)(b * SEQ + s)) * HIDDEN + d] = octx[j][r] / lrow[r];
        }
    }
}

extern "C" void kernel_launch(void* const* d_in, const int* in_sizes, int n_in,
                              void* d_out, int out_size, void* d_ws, size_t ws_size,
                              hipStream_t stream) {
    const float* X  = (const float*)d_in[0];
    const float* Wq = (const float*)d_in[1];
    const float* Wk = (const float*)d_in[2];
    const float* Wv = (const float*)d_in[3];
    float* out = (float*)d_out;

    char* ws = (char*)d_ws;
    half_t* Xh  = (half_t*)(ws + XH_OFF);
    half_t* Wqh = (half_t*)(ws + WQH_OFF);
    half_t* Wkh = (half_t*)(ws + WKH_OFF);
    half_t* Wvh = (half_t*)(ws + WVH_OFF);
    half_t* Qh  = (half_t*)(ws + QH_OFF);
    half_t* Kh  = (half_t*)(ws + KH_OFF);
    half_t* Vt  = (half_t*)(ws + VT_OFF);
    float*  cosT = (float*)(ws + COS_OFF);
    float*  sinT = (float*)(ws + SIN_OFF);

    // 1) precision convert
    {
        int nX = MTOT * HIDDEN;
        int nW = HIDDEN * HIDDEN;
        cvt_f32_f16<<<(nX + 255) / 256, 256, 0, stream>>>(X,  Xh,  nX);
        cvt_f32_f16<<<(nW + 255) / 256, 256, 0, stream>>>(Wq, Wqh, nW);
        cvt_f32_f16<<<(nW + 255) / 256, 256, 0, stream>>>(Wk, Wkh, nW);
        cvt_f32_f16<<<(nW + 255) / 256, 256, 0, stream>>>(Wv, Wvh, nW);
    }
    // 2) RoPE tables
    {
        int n = SEQ * HEAD_DIM;
        rope_tables<<<(n + 255) / 256, 256, 0, stream>>>(cosT, sinT);
    }
    // 3) QKV projections (+RoPE for Q/K, transpose-store for V)
    {
        dim3 grid(MTOT / 128, HIDDEN / 64);   // (32, 16)
        qkv_gemm<<<grid, 256, 0, stream>>>(Xh, Wqh, cosT, sinT, Qh, nullptr, 1);
        qkv_gemm<<<grid, 256, 0, stream>>>(Xh, Wkh, cosT, sinT, Kh, nullptr, 1);
        qkv_gemm<<<grid, 256, 0, stream>>>(Xh, Wvh, cosT, sinT, nullptr, Vt, 0);
    }
    // 4) flash attention
    {
        dim3 grid(SEQ / 16 / 4, BATCH * NUM_HEADS);  // (32, 32)
        flash_attn<<<grid, 128, 0, stream>>>(Qh, Kh, Vt, out);
    }
    (void)in_sizes; (void)n_in; (void)out_size; (void)ws_size;
}
